// TDMHSA_11381663334495
// MI455X (gfx1250) — matrix operand order repaired and checked by reading, not compiled, not run on hardware
//
#include <hip/hip_runtime.h>
#include <hip/hip_bf16.h>

// ---------------------------------------------------------------------------
// Relative-position MHSA with time bias, CDNA5 (gfx1250) wave32 + WMMA bf16
// + async global->LDS DMA (ASYNCcnt) with double-buffered K/V tiles.
// B=8 S=1024 D=768 H=12 HD=64 T=384, SCALE = 64^-0.5 = 0.125
// ---------------------------------------------------------------------------

typedef __bf16 bf16;
typedef __bf16 v8bf  __attribute__((ext_vector_type(8)));
typedef __bf16 v16bf __attribute__((ext_vector_type(16)));
typedef float  v8f   __attribute__((ext_vector_type(8)));
typedef int    v4i   __attribute__((vector_size(16)));   // matches builtin proto

#if __has_builtin(__builtin_amdgcn_global_load_async_to_lds_b128)
#define HAVE_ASYNC 1
#else
#define HAVE_ASYNC 0
#endif

// builtin proto: (v4i AS(1)* gsrc, v4i AS(3)* ldst, imm offset, imm cpol)
#define ASG4(p) ((__attribute__((address_space(1))) v4i*)(p))
#define ASL4(p) ((__attribute__((address_space(3))) v4i*)(p))

__device__ __forceinline__ void wait_async0() {
#if __has_builtin(__builtin_amdgcn_s_wait_asynccnt)
  __builtin_amdgcn_s_wait_asynccnt(0);
#elif HAVE_ASYNC
  asm volatile("s_wait_asynccnt 0x0" ::: "memory");
#endif
}

__device__ __forceinline__ v16bf cat(v8bf lo, v8bf hi) {
  return __builtin_shufflevector(lo, hi, 0,1,2,3,4,5,6,7,8,9,10,11,12,13,14,15);
}

__device__ __forceinline__ unsigned short f2bfbits(float x) {
  union { float f; unsigned u; } c; c.f = x;
  unsigned r = c.u + 0x7FFFu + ((c.u >> 16) & 1u);   // round-to-nearest-even
  return (unsigned short)(r >> 16);
}
__device__ __forceinline__ bf16 f2bf(float x) {
  union { unsigned short u; bf16 b; } o; o.u = f2bfbits(x);
  return o.b;
}

__device__ __forceinline__ v8f wmma_bf16(v16bf a, v16bf b, v8f c) {
  // (neg_a, A, neg_b, B, c_mod, C, reuse_a, reuse_b)
  return __builtin_amdgcn_wmma_f32_16x16x32_bf16(false, a, false, b, (short)0, c,
                                                 false, false);
}

// --------------------------- prep kernels ----------------------------------

__global__ void k_cvt(const float* __restrict__ x, bf16* __restrict__ y, int n) {
  for (int i = blockIdx.x * blockDim.x + threadIdx.x; i < n;
       i += gridDim.x * blockDim.x)
    y[i] = f2bf(x[i]);
}

// Wt[n][k] = W[k][n], 768x768, f32 -> bf16
__global__ void k_wt(const float* __restrict__ w, bf16* __restrict__ wt) {
  int i = blockIdx.x * blockDim.x + threadIdx.x;
  if (i >= 768 * 768) return;
  int n = i / 768, k = i % 768;
  wt[n * 768 + k] = f2bf(w[k * 768 + n]);
}

// tproj[b][n] = sum_t time_emb[b][t] * W[t][n] + bias[n]   (8x768, tiny)
__global__ void k_tproj(const float* __restrict__ te, const float* __restrict__ W,
                        const float* __restrict__ bias, float* __restrict__ out) {
  int i = blockIdx.x * blockDim.x + threadIdx.x;
  if (i >= 8 * 768) return;
  int b = i / 768, n = i % 768;
  float s = bias[n];
  for (int t = 0; t < 384; ++t) s += te[b * 384 + t] * W[t * 768 + n];
  out[i] = s;
}

// --------------------------- projection GEMM -------------------------------
// out = patches_bf @ Wt^T (+bias +tproj), M=8192, N=768, K=768.
// mode 0: out0=q (qs+qt), out1=qs  ; mode 1: out0=k ; mode 2: out0=v^T
// Plain fully-unrolled k-loop: unrolled straight-line code lets the scheduler
// hoist global_load_b128 clauses across WMMAs (manual pipelining de-unrolls
// the loop and generates v_mov rotation storms instead).
__global__ __launch_bounds__(128) void k_proj(
    const bf16* __restrict__ A, const bf16* __restrict__ Wt,
    const float* __restrict__ bias, const float* __restrict__ tp,
    bf16* __restrict__ out0, bf16* __restrict__ out1, int mode) {
  int m0 = blockIdx.x * 64, n0 = blockIdx.y * 64;
  int lane = threadIdx.x & 31, w = threadIdx.x >> 5;
  int l16 = lane & 15, half = lane >> 4;
  int mrow = m0 + 16 * w + l16;  // A-fragment row for this lane

  const bf16* arow = A + mrow * 768;
  v8f acc[4] = {};
#pragma unroll
  for (int k0 = 0; k0 < 768; k0 += 32) {
    const bf16* ap = arow + k0 + 16 * half;
    v16bf af = cat(*(const v8bf*)ap, *(const v8bf*)(ap + 8));
#pragma unroll
    for (int t = 0; t < 4; ++t) {
      const bf16* bp = Wt + (n0 + 16 * t + l16) * 768 + k0 + 16 * half;
      v16bf bfr = cat(*(const v8bf*)bp, *(const v8bf*)(bp + 8));
      acc[t] = wmma_bf16(af, bfr, acc[t]);
    }
  }

  int b = m0 >> 10;  // 64-row tile never crosses a batch boundary (1024%64==0)
#pragma unroll
  for (int t = 0; t < 4; ++t) {
    int n = n0 + 16 * t + l16;
    int h = n >> 6, hd = n & 63;
    float bi = bias[n], tpv = tp[b * 768 + n];
#pragma unroll
    for (int r = 0; r < 8; ++r) {
      int m = m0 + 16 * w + 8 * half + r;
      int s = m & 1023;
      float v = acc[t][r] + bi;
      long base = ((long)(b * 12 + h) * 1024 + s) * 64 + hd;
      if (mode == 0) {
        out1[base] = f2bf(v);               // qs (no time bias; used for QEr)
        out0[base] = f2bf(v + tpv);         // q = qs + qt
      } else if (mode == 1) {
        out0[base] = f2bf(v + tpv);         // k
      } else {
        out0[((long)(b * 12 + h) * 64 + hd) * 1024 + s] = f2bf(v + tpv);  // v^T
      }
    }
  }
}

// --------------------------- fused attention -------------------------------
// One workgroup (4 waves, 128 thr) per (b, h, 64-query tile).
// LDS: G panel 65 x 1028 f32, double-buffered K (64x64) / V^T (64x64) bf16
// tiles filled by async global->LDS DMA, P staging 4 x 16x64 bf16.
constexpr int GSTR   = 1028;
constexpr int LDS_G  = 65 * GSTR * 4;            // 267280
constexpr int LDS_KV = 64 * 64 * 2;              // 8192
constexpr int OFF_K0 = LDS_G;                    // layout: K0 V0 K1 V1 P
constexpr int OFF_V0 = OFF_K0 + LDS_KV;
constexpr int OFF_P  = OFF_K0 + 4 * LDS_KV;
constexpr int LDS_SZ = OFF_P + 4 * 16 * 64 * 2;  // 308240 <= 320KB WGP LDS

// Issue one 64x64 K tile + 64x64 V^T tile into LDS (async DMA when available).
__device__ __forceinline__ void issue_tile(const bf16* __restrict__ kp,
                                           const bf16* __restrict__ vp, int n0,
                                           bf16* Kd, bf16* Vd, int tid) {
#if HAVE_ASYNC
#pragma unroll
  for (int j = 0; j < 4; ++j) {
    int v = tid + 128 * j;                     // 512 16-byte vectors per tile
    const bf16* ks = kp + n0 * 64 + v * 8;     // K tile is contiguous
    __builtin_amdgcn_global_load_async_to_lds_b128(ASG4(ks), ASL4(Kd + v * 8), 0, 0);
    int hd = v >> 3, c8 = v & 7;
    const bf16* vs = vp + hd * 1024 + n0 + c8 * 8;
    __builtin_amdgcn_global_load_async_to_lds_b128(ASG4(vs), ASL4(Vd + v * 8), 0, 0);
  }
#else
  const uint4* ksrc = (const uint4*)(kp + n0 * 64);
  uint4* kdst = (uint4*)Kd;
  uint4* vdst = (uint4*)Vd;
  for (int v = tid; v < 512; v += 128) {
    kdst[v] = ksrc[v];
    int hd = v >> 3, c8 = v & 7;
    vdst[v] = *(const uint4*)(vp + hd * 1024 + n0 + c8 * 8);
  }
#endif
}

__global__ __launch_bounds__(128) void k_attn(
    const bf16* __restrict__ Q, const bf16* __restrict__ Qs,
    const bf16* __restrict__ K, const bf16* __restrict__ Vt,
    const bf16* __restrict__ Er, float* __restrict__ out) {
  extern __shared__ char smem[];
  float* G = (float*)smem;
  unsigned short* P = (unsigned short*)(smem + OFF_P);
  // double-buffer addresses computed arithmetically (no static pointer arrays:
  // addrspacecast of the LDS symbol in a static initializer breaks lld)
  auto Kbuf = [&](int p) { return (bf16*)(smem + OFF_K0 + p * (2 * LDS_KV)); };
  auto Vbuf = [&](int p) { return (bf16*)(smem + OFF_V0 + p * (2 * LDS_KV)); };

  int m0 = blockIdx.x * 64;
  int h = blockIdx.y, b = blockIdx.z;
  int tid = threadIdx.x;
  int lane = tid & 31, w = tid >> 5;
  int l16 = lane & 15, half = lane >> 4;

  const bf16* qsp = Qs + (long)(b * 12 + h) * 1024 * 64;
  const bf16* qp  = Q  + (long)(b * 12 + h) * 1024 * 64;
  const bf16* kp  = K  + (long)(b * 12 + h) * 1024 * 64;
  const bf16* vp  = Vt + (long)(b * 12 + h) * 64 * 1024;

  // kick off DMA of the first K/V tile; it overlaps the G-panel WMMAs below
  issue_tile(kp, vp, 0, Kbuf(0), Vbuf(0), tid);

  // ---- Phase 1: G[i][e] = qs[m0+i] . Er[e],  i in [0,64], e in [0,1023] ----
  for (int tile = w; tile < 5 * 64; tile += 4) {
    int mi = tile % 5, ei = tile / 5;
    int row = m0 + 16 * mi + l16;
    if (row > 1023) row = 1023;                 // clamp (clamped rows unused)
    v8f c = {};
#pragma unroll
    for (int k0 = 0; k0 < 64; k0 += 32) {
      const bf16* ap = qsp + row * 64 + k0 + 16 * half;
      v16bf af = cat(*(const v8bf*)ap, *(const v8bf*)(ap + 8));
      const bf16* bp = Er + (16 * ei + l16) * 64 + k0 + 16 * half;
      v16bf bfr = cat(*(const v8bf*)bp, *(const v8bf*)(bp + 8));
      c = wmma_bf16(af, bfr, c);
    }
#pragma unroll
    for (int r = 0; r < 8; ++r) {
      int i = 16 * mi + 8 * half + r;
      if (i <= 64) G[i * GSTR + 16 * ei + l16] = c[r];
    }
  }
  __syncthreads();

  // ---- Phase 2: flash attention, each wave owns 16 full query rows ----
  int qrow = m0 + 16 * w + l16;
  v16bf qf[2];
#pragma unroll
  for (int k0 = 0; k0 < 64; k0 += 32) {
    const bf16* ap = qp + qrow * 64 + k0 + 16 * half;
    qf[k0 / 32] = cat(*(const v8bf*)ap, *(const v8bf*)(ap + 8));
  }

  v8f o[4] = {};
  float rmax[8], rsum[8];
#pragma unroll
  for (int r = 0; r < 8; ++r) { rmax[r] = -1e30f; rsum[r] = 0.f; }
  unsigned short* Pw = P + w * 16 * 64;

  for (int kt = 0; kt < 16; ++kt) {
    int n0 = kt * 64;
    wait_async0();            // this wave's outstanding tile DMA complete
    __syncthreads();          // all waves' DMA complete; prev compute done
    if (kt < 15)
      issue_tile(kp, vp, n0 + 64, Kbuf((kt + 1) & 1), Vbuf((kt + 1) & 1), tid);
    const bf16* Kt = Kbuf(kt & 1);
    const bf16* Vs = Vbuf(kt & 1);

    // scores = Q . K^T
    v8f sc[4];
#pragma unroll
    for (int t = 0; t < 4; ++t) {
      v8f c = {};
#pragma unroll
      for (int k0 = 0; k0 < 64; k0 += 32) {
        const bf16* bp = Kt + (16 * t + l16) * 64 + k0 + 16 * half;
        v16bf bfr = cat(*(const v8bf*)bp, *(const v8bf*)(bp + 8));
        c = wmma_bf16(qf[k0 / 32], bfr, c);
      }
      sc[t] = c;
    }

    // += Srel (skewed gather from G panel), then scale
#pragma unroll
    for (int t = 0; t < 4; ++t) {
#pragma unroll
      for (int r = 0; r < 8; ++r) {
        int ml = 16 * w + 8 * half + r;          // local query row 0..63
        int n = n0 + 16 * t + l16;
        int rel = n - (m0 + ml);
        float srel;
        if (rel <= 0)       srel = G[ml * GSTR + 1023 + rel];
        else if (rel == 1)  srel = 0.f;
        else                srel = G[(ml + 1) * GSTR + rel - 2];
        sc[t][r] = (sc[t][r] + srel) * 0.125f;
      }
    }

    // online softmax (rows live in 16-lane groups -> shfl_xor masks 1..8)
    float alpha[8];
#pragma unroll
    for (int r = 0; r < 8; ++r) {
      float mx = fmaxf(fmaxf(sc[0][r], sc[1][r]), fmaxf(sc[2][r], sc[3][r]));
#pragma unroll
      for (int msk = 1; msk < 16; msk <<= 1) mx = fmaxf(mx, __shfl_xor(mx, msk, 32));
      float nm = fmaxf(rmax[r], mx);
      alpha[r] = __expf(rmax[r] - nm);
      rmax[r] = nm;
      float ps = 0.f;
#pragma unroll
      for (int t = 0; t < 4; ++t) {
        float p = __expf(sc[t][r] - nm);
        sc[t][r] = p;
        ps += p;
      }
#pragma unroll
      for (int msk = 1; msk < 16; msk <<= 1) ps += __shfl_xor(ps, msk, 32);
      rsum[r] = rsum[r] * alpha[r] + ps;
    }
#pragma unroll
    for (int t = 0; t < 4; ++t)
#pragma unroll
      for (int r = 0; r < 8; ++r) o[t][r] *= alpha[r];

    // re-layout P: C-fragment -> LDS -> A-fragment (bf16)
#pragma unroll
    for (int t = 0; t < 4; ++t)
#pragma unroll
      for (int r = 0; r < 8; ++r)
        Pw[(8 * half + r) * 64 + 16 * t + l16] = f2bfbits(sc[t][r]);

    // O += P @ V
#pragma unroll
    for (int k0 = 0; k0 < 64; k0 += 32) {
      const bf16* pp = (const bf16*)Pw + l16 * 64 + k0 + 16 * half;
      v16bf pf = cat(*(const v8bf*)pp, *(const v8bf*)(pp + 8));
#pragma unroll
      for (int t = 0; t < 4; ++t) {
        const bf16* bp = Vs + (16 * t + l16) * 64 + k0 + 16 * half;
        v16bf vf = cat(*(const v8bf*)bp, *(const v8bf*)(bp + 8));
        o[t] = wmma_bf16(pf, vf, o[t]);
      }
    }
  }

  // epilogue: out[b][s][h*64+hd] = O / l
#pragma unroll
  for (int t = 0; t < 4; ++t) {
#pragma unroll
    for (int r = 0; r < 8; ++r) {
      int m = m0 + 16 * w + 8 * half + r;
      int hd = 16 * t + l16;
      out[((long)b * 1024 + m) * 768 + h * 64 + hd] = o[t][r] / rsum[r];
    }
  }
}

// --------------------------- launcher --------------------------------------

extern "C" void kernel_launch(void* const* d_in, const int* in_sizes, int n_in,
                              void* d_out, int out_size, void* d_ws, size_t ws_size,
                              hipStream_t stream) {
  const float* patches  = (const float*)d_in[0];
  const float* time_emb = (const float*)d_in[1];
  const float* Wqs = (const float*)d_in[2];  const float* bqs = (const float*)d_in[3];
  const float* Wks = (const float*)d_in[4];  const float* bks = (const float*)d_in[5];
  const float* Wvs = (const float*)d_in[6];  const float* bvs = (const float*)d_in[7];
  const float* Wqt = (const float*)d_in[8];  const float* bqt = (const float*)d_in[9];
  const float* Wkt = (const float*)d_in[10]; const float* bkt = (const float*)d_in[11];
  const float* Wvt = (const float*)d_in[12]; const float* bvt = (const float*)d_in[13];
  const float* Er  = (const float*)d_in[14];
  float* outp = (float*)d_out;

  // workspace partition (~64 MB)
  char* ws = (char*)d_ws;
  bf16* patches_bf = (bf16*)ws;          ws += (size_t)8192 * 768 * 2;
  bf16* WtQ = (bf16*)ws;                 ws += (size_t)768 * 768 * 2;
  bf16* WtK = (bf16*)ws;                 ws += (size_t)768 * 768 * 2;
  bf16* WtV = (bf16*)ws;                 ws += (size_t)768 * 768 * 2;
  bf16* Erb = (bf16*)ws;                 ws += (size_t)1024 * 64 * 2;
  float* tq = (float*)ws;                ws += (size_t)8 * 768 * 4;
  float* tk = (float*)ws;                ws += (size_t)8 * 768 * 4;
  float* tv = (float*)ws;                ws += (size_t)8 * 768 * 4;
  bf16* q_bf  = (bf16*)ws;               ws += (size_t)8 * 12 * 1024 * 64 * 2;
  bf16* qs_bf = (bf16*)ws;               ws += (size_t)8 * 12 * 1024 * 64 * 2;
  bf16* k_bf  = (bf16*)ws;               ws += (size_t)8 * 12 * 1024 * 64 * 2;
  bf16* vt_bf = (bf16*)ws;               ws += (size_t)8 * 12 * 1024 * 64 * 2;

  k_cvt<<<4096, 256, 0, stream>>>(patches, patches_bf, 8192 * 768);
  k_cvt<<<256, 256, 0, stream>>>(Er, Erb, 1024 * 64);
  k_wt<<<(768 * 768 + 255) / 256, 256, 0, stream>>>(Wqs, WtQ);
  k_wt<<<(768 * 768 + 255) / 256, 256, 0, stream>>>(Wks, WtK);
  k_wt<<<(768 * 768 + 255) / 256, 256, 0, stream>>>(Wvs, WtV);
  k_tproj<<<(8 * 768 + 255) / 256, 256, 0, stream>>>(time_emb, Wqt, bqt, tq);
  k_tproj<<<(8 * 768 + 255) / 256, 256, 0, stream>>>(time_emb, Wkt, bkt, tk);
  k_tproj<<<(8 * 768 + 255) / 256, 256, 0, stream>>>(time_emb, Wvt, bvt, tv);

  dim3 gp(128, 12);
  k_proj<<<gp, 128, 0, stream>>>(patches_bf, WtQ, bqs, tq, q_bf, qs_bf, 0);
  k_proj<<<gp, 128, 0, stream>>>(patches_bf, WtK, bks, tk, k_bf, nullptr, 1);
  k_proj<<<gp, 128, 0, stream>>>(patches_bf, WtV, bvs, tv, vt_bf, nullptr, 2);

  dim3 ga(16, 12, 8);
  k_attn<<<ga, 128, LDS_SZ, stream>>>(q_bf, qs_bf, k_bf, vt_bf, Erb, outp);
}
